// VectorQuantizer_84911503441993
// MI455X (gfx1250) — matrix operand order repaired and checked by reading, not compiled, not run on hardware
//
#include <hip/hip_runtime.h>
#include <stdint.h>

// ---------------------------------------------------------------------------
// VectorQuantizer for MI455X (gfx1250, wave32, WMMA).
// dist argmin:  argmin_k ||c_k||^2 - 2 z.c_k   (||z||^2 dropped: row-constant)
// Cross term via v_wmma_f32_16x16x32_bf16; c_sq exact in f32.
// B resident for TWO 16-column tiles (256 VGPRs, uses CDNA5 >256-VGPR file):
// each A fragment from LDS feeds 2 WMMAs -> halves LDS stream traffic.
// ---------------------------------------------------------------------------

typedef __attribute__((ext_vector_type(16))) __bf16 v16bf;
typedef __attribute__((ext_vector_type(8)))  float  v8f;

#define D_DIM          512
#define K_CODES        512
#define ROWS_PER_BLOCK 128
#define THREADS        256
#define LDS_STRIDE     520                      // bf16 elems per LDS row (512 + 8 pad)
#define LDS_A_BYTES    (ROWS_PER_BLOCK * LDS_STRIDE * 2)   // 133120
#define LDS_TOTAL      (LDS_A_BYTES + ROWS_PER_BLOCK * 8)  // + minbuf[128] u64

__device__ __forceinline__ uint32_t f32_to_bf16_rne(float f) {
  uint32_t u = __float_as_uint(f);
  uint32_t r = u + 0x7FFFu + ((u >> 16) & 1u);   // round-to-nearest-even
  return r >> 16;
}
__device__ __forceinline__ uint32_t pack2bf(float a, float b) {
  return f32_to_bf16_rne(a) | (f32_to_bf16_rne(b) << 16);
}

union ABFrag { uint4 u[2]; v16bf v; };

// ------------------------- prep: c_sq[k] = ||codebook[k]||^2 ----------------
__global__ void vq_csq_kernel(const float* __restrict__ cb, float* __restrict__ c_sq) {
  __shared__ float red[64];
  const int k   = blockIdx.x;
  const int tid = threadIdx.x;
  const float4* row = (const float4*)(cb + (size_t)k * D_DIM);
  float s = 0.f;
  for (int i = tid; i < D_DIM / 4; i += 64) {
    float4 v = row[i];
    s += v.x * v.x + v.y * v.y + v.z * v.z + v.w * v.w;
  }
  red[tid] = s;
  __syncthreads();
  for (int off = 32; off > 0; off >>= 1) {
    if (tid < off) red[tid] += red[tid + off];
    __syncthreads();
  }
  if (tid == 0) c_sq[k] = red[0];
}

// ------------------------------- main kernel --------------------------------
__global__ void __launch_bounds__(THREADS)
vq_main_kernel(const float* __restrict__ z, const float* __restrict__ cb,
               const float* __restrict__ c_sq, float* __restrict__ out) {
  extern __shared__ char smem[];
  unsigned short*     ldsA   = (unsigned short*)smem;                 // [128][520] bf16
  unsigned long long* minbuf = (unsigned long long*)(smem + LDS_A_BYTES);

  const int    tid     = threadIdx.x;
  const int    lane    = tid & 31;
  const int    wave    = tid >> 5;          // 0..7
  const int    l15     = lane & 15;
  const int    halfsel = lane >> 4;         // 0 or 1
  const size_t rowBase = (size_t)blockIdx.x * ROWS_PER_BLOCK;

  if (tid < ROWS_PER_BLOCK) minbuf[tid] = ~0ull;

  // ---- stage z tile (128 x 512 f32) -> LDS bf16, padded rows ----
  {
    const float4* zb = (const float4*)(z + rowBase * D_DIM);
#pragma unroll 4
    for (int i = 0; i < 64; ++i) {
      int    j   = tid + i * THREADS;       // float4 index, 0..16383
      float4 v   = zb[j];
      int    row = j >> 7;                  // 128 float4 per row
      int    dco = (j & 127) << 2;          // element offset within row
      uint2  st;
      st.x = pack2bf(v.x, v.y);
      st.y = pack2bf(v.z, v.w);
      *(uint2*)(ldsA + row * LDS_STRIDE + dco) = st;   // ds_store_b64
    }
  }
  __syncthreads();

  // ---- GEMM + running argmin.  Wave w owns columns [w*64, w*64+64),
  //      processed as 2 pairs of 16-column tiles with B resident for both. ----
  for (int pair = 0; pair < 2; ++pair) {
    const int pbase = wave * 64 + pair * 32;

    // B fragments for BOTH tiles of the pair, all 16 k-steps, in VGPRs (256).
    // Lane (l15, halfsel) supplies B[K = kk*32 + halfsel*16 + j][N = l15].
    v16bf Bfr[2][16];
    float csq_l[2];
#pragma unroll
    for (int t = 0; t < 2; ++t) {
      const int col = pbase + t * 16 + l15;
#pragma unroll
      for (int kk = 0; kk < 16; ++kk) {
        const float4* p = (const float4*)(cb + (size_t)col * D_DIM + kk * 32 + halfsel * 16);
        float4 f0 = p[0], f1 = p[1], f2 = p[2], f3 = p[3];
        ABFrag b;
        b.u[0].x = pack2bf(f0.x, f0.y); b.u[0].y = pack2bf(f0.z, f0.w);
        b.u[0].z = pack2bf(f1.x, f1.y); b.u[0].w = pack2bf(f1.z, f1.w);
        b.u[1].x = pack2bf(f2.x, f2.y); b.u[1].y = pack2bf(f2.z, f2.w);
        b.u[1].z = pack2bf(f3.x, f3.y); b.u[1].w = pack2bf(f3.z, f3.w);
        Bfr[t][kk] = b.v;
      }
      csq_l[t] = c_sq[col];
    }

    for (int mt = 0; mt < 8; ++mt) {
      v8f acc0 = {0.f, 0.f, 0.f, 0.f, 0.f, 0.f, 0.f, 0.f};
      v8f acc1 = {0.f, 0.f, 0.f, 0.f, 0.f, 0.f, 0.f, 0.f};
      // A fragment source: row = mt*16 + l15; lanes<16 take K {0..15},
      // lanes>=16 take K {16..31}  (contiguous order, matches B).
      const unsigned short* aptr = ldsA + (mt * 16 + l15) * LDS_STRIDE + halfsel * 16;

#pragma unroll
      for (int kk = 0; kk < 16; ++kk) {
        ABFrag a;
        a.u[0] = *(const uint4*)(aptr + kk * 32);        // ds_load_b128
        a.u[1] = *(const uint4*)(aptr + kk * 32 + 8);    // ds_load_b128
        // One A fragment feeds both resident B tiles: 2 WMMAs per LDS trip.
        acc0 = __builtin_amdgcn_wmma_f32_16x16x32_bf16(
                   false, a.v, false, Bfr[0][kk], (short)0, acc0, false, false);
        acc1 = __builtin_amdgcn_wmma_f32_16x16x32_bf16(
                   false, a.v, false, Bfr[1][kk], (short)0, acc1, false, false);
      }

      // acc[r]: row = mt*16 + r + halfsel*8, col = pbase + t*16 + l15
#pragma unroll
      for (int t = 0; t < 2; ++t) {
#pragma unroll
        for (int r = 0; r < 8; ++r) {
          float av  = (t == 0) ? acc0[r] : acc1[r];
          float val = csq_l[t] - 2.0f * av;
          int   idx = pbase + t * 16 + l15;
#pragma unroll
          for (int m = 1; m < 16; m <<= 1) {             // reduce 16-lane half
            float ov = __shfl_xor(val, m, 32);
            int   oi = __shfl_xor(idx, m, 32);
            if (ov < val || (ov == val && oi < idx)) { val = ov; idx = oi; }
          }
          if (l15 == 0) {
            int      row = mt * 16 + r + halfsel * 8;
            uint32_t b   = __float_as_uint(val);
            uint32_t key = (b & 0x80000000u) ? ~b : (b | 0x80000000u);
            unsigned long long packed = ((unsigned long long)key << 32) | (unsigned)idx;
            atomicMin(&minbuf[row], packed);             // ds_min_u64
          }
        }
      }
    }
  }
  __syncthreads();

  // ---- gather winning codebook rows -> out (2 threads per row, float4) ----
  {
    const int      orow = tid >> 1;
    const int      half = tid & 1;
    const unsigned idx  = (unsigned)(minbuf[orow] & 0xFFFFFFFFu);
    const float4* src = (const float4*)(cb + (size_t)idx * D_DIM + half * 256);
    float4*       dst = (float4*)(out + (rowBase + orow) * D_DIM + half * 256);
#pragma unroll 4
    for (int i = 0; i < 64; ++i) dst[i] = src[i];
  }
}

// ------------------------------- launcher -----------------------------------
extern "C" void kernel_launch(void* const* d_in, const int* in_sizes, int n_in,
                              void* d_out, int out_size, void* d_ws, size_t ws_size,
                              hipStream_t stream) {
  (void)in_sizes; (void)n_in; (void)out_size; (void)ws_size;
  const float* z    = (const float*)d_in[0];   // [32,4096,512] f32
  const float* cb   = (const float*)d_in[1];   // [512,512] f32
  float*       csq  = (float*)d_ws;            // 512 f32 scratch
  float*       out  = (float*)d_out;           // [32,4096,512] f32

  vq_csq_kernel<<<K_CODES, 64, 0, stream>>>(cb, csq);

  (void)hipFuncSetAttribute((const void*)vq_main_kernel,
                            hipFuncAttributeMaxDynamicSharedMemorySize,
                            (int)LDS_TOTAL);
  const int nblocks = (32 * 4096) / ROWS_PER_BLOCK;     // 1024
  vq_main_kernel<<<nblocks, THREADS, LDS_TOTAL, stream>>>(z, cb, csq, out);
}
